// GPT2Attention_84559316123761
// MI455X (gfx1250) — matrix operand order repaired and checked by reading, not compiled, not run on hardware
//
#include <hip/hip_runtime.h>

typedef __attribute__((ext_vector_type(16))) __bf16 v16bf;
typedef __attribute__((ext_vector_type(8)))  __bf16 v8bf;
typedef __attribute__((ext_vector_type(8)))  float  v8f;
typedef __attribute__((ext_vector_type(4)))  float  v4f;
typedef __attribute__((ext_vector_type(4)))  unsigned v4u;

// B=2, S=2048, E=1024, H=16, D=64
#define S_LEN 2048
#define E_DIM 1024
#define H_NUM 16
#define D_DIM 64

// Packed f32->bf16 conversion on the VALU (1 instr per 2 values).
static __device__ __forceinline__ unsigned cvt_pk_bf16(float lo, float hi) {
  unsigned r;
  asm("v_cvt_pk_bf16_f32 %0, %1, %2" : "=v"(r) : "v"(lo), "v"(hi));
  return r;
}
static __device__ __forceinline__ __bf16 to_bf16_hw(float f) {
  unsigned r = cvt_pk_bf16(f, f);
  union { unsigned u; __bf16 b[2]; } c; c.u = r; return c.b[0];
}

// LDS byte offset of a __shared__ object: LDS flat addresses keep the
// workgroup-LDS offset in the low 32 bits (aperture base is in the high bits).
static __device__ __forceinline__ unsigned lds_offset(const void* p) {
  return (unsigned)(unsigned long long)p;
}

union V16U { v16bf v; v8bf h[2]; };

// ---------------------------------------------------------------------------
// Kernel 1: QKV = X @ W_attn + b_attn, scattered into bf16 Q[B,H,S,D],
// K[B,H,S,D], Vt[B,H,D,S].  M=4096, N=3072, K=1024.  Tile 128x64, BK=32.
// LDS is written in WMMA-fragment order: each lane's fragment is one
// contiguous 32B chunk.
// ---------------------------------------------------------------------------
__global__ __launch_bounds__(256)
void qkv_gemm_kernel(const float* __restrict__ X, const float* __restrict__ W,
                     const float* __restrict__ bias,
                     __bf16* __restrict__ Qo, __bf16* __restrict__ Ko,
                     __bf16* __restrict__ Vto) {
  const int K = 1024, N = 3072;
  __shared__ __align__(32) __bf16 As[8][32][16];  // [rowgroup][lane][16 bf16]
  __shared__ __align__(32) __bf16 Bs[4][32][16];  // [colgroup][lane][16 bf16]

  const int tid = threadIdx.x, lane = tid & 31, wv = tid >> 5;
  const int m0 = blockIdx.y * 128, n0 = blockIdx.x * 64;
  const int wrg = (wv >> 1) * 2;      // row-groups wrg, wrg+1
  const int wcg = (wv & 1) * 2;       // col-groups wcg, wcg+1

  v8f acc[2][2];
#pragma unroll
  for (int i = 0; i < 2; ++i)
#pragma unroll
    for (int j = 0; j < 2; ++j)
#pragma unroll
      for (int e = 0; e < 8; ++e) acc[i][j][e] = 0.0f;

  for (int kt = 0; kt < K; kt += 32) {
    __syncthreads();
    // ---- stage A: 128x32 fp32 -> bf16 fragment-order LDS ----
#pragma unroll
    for (int t = 0; t < 2; ++t) {
      int task = tid + t * 256;          // 512 tasks of 8 elements
      int mloc = task >> 2;
      int kc   = (task & 3) << 3;
      const float* src = X + (size_t)(m0 + mloc) * K + kt + kc;
      v4f a0 = *(const v4f*)src;
      v4f a1 = *(const v4f*)(src + 4);
      if (kt + 32 < K) __builtin_prefetch(src + 32);   // global_prefetch_b8
      int g = mloc >> 4, row = mloc & 15;
      int lane_s = (((kc >> 3) & 1) << 4) + row;       // half*16 + row
      v4u w;
      w[0] = cvt_pk_bf16(a0[0], a0[1]);
      w[1] = cvt_pk_bf16(a0[2], a0[3]);
      w[2] = cvt_pk_bf16(a1[0], a1[1]);
      w[3] = cvt_pk_bf16(a1[2], a1[3]);
      *(v4u*)&As[g][lane_s][(kc >> 4) << 3] = w;
    }
    // ---- stage B: 32x64 fp32 (strided columns) -> fragment-order LDS ----
    {
      int ncol = tid & 63, kq = tid >> 6;   // kq in 0..3 covers 8 k each
      const float* wsrc = W + (size_t)(kt + (kq << 3)) * N + n0 + ncol;
      int lane_s = ((kq & 1) << 4) + (ncol & 15);
      int cg = ncol >> 4, off = (kq >> 1) << 3;
      float f[8];
#pragma unroll
      for (int j2 = 0; j2 < 8; ++j2) f[j2] = wsrc[(size_t)j2 * N];
      v4u w;
#pragma unroll
      for (int j2 = 0; j2 < 4; ++j2) w[j2] = cvt_pk_bf16(f[2 * j2], f[2 * j2 + 1]);
      *(v4u*)&Bs[cg][lane_s][off] = w;
    }
    __syncthreads();
    // ---- compute: 4 WMMA per wave per K-step ----
    v16bf af[2], bfr[2];
    af[0]  = *(const v16bf*)&As[wrg][lane][0];
    af[1]  = *(const v16bf*)&As[wrg + 1][lane][0];
    bfr[0] = *(const v16bf*)&Bs[wcg][lane][0];
    bfr[1] = *(const v16bf*)&Bs[wcg + 1][lane][0];
#pragma unroll
    for (int i = 0; i < 2; ++i)
#pragma unroll
      for (int j = 0; j < 2; ++j)
        acc[i][j] = __builtin_amdgcn_wmma_f32_16x16x32_bf16(
            false, af[i], false, bfr[j], (short)0, acc[i][j], false, false);
  }

  // ---- epilogue: bias + scatter.  Region (Q/K/V) is block-uniform since
  // the 64-wide N tile never straddles an E_DIM boundary. ----
  const int rowbase = (lane >> 4) << 3;   // C layout: rows r or r+8
  const int ncl = lane & 15;
  const int region = n0 >> 10;            // 0=Q, 1=K, 2=V  (scalar)
  __bf16* __restrict__ dstQK = (region == 0) ? Qo : Ko;
#pragma unroll
  for (int i = 0; i < 2; ++i)
#pragma unroll
    for (int j = 0; j < 2; ++j)
#pragma unroll
      for (int r = 0; r < 8; ++r) {
        int mg = m0 + ((wrg + i) << 4) + r + rowbase;
        int ng = n0 + ((wcg + j) << 4) + ncl;
        float v = acc[i][j][r] + bias[ng];
        int bidx = mg >> 11, s = mg & (S_LEN - 1);
        int nr = ng & (E_DIM - 1);
        int h = nr >> 6, d = nr & 63;
        if (region < 2) {
          dstQK[(((size_t)bidx * H_NUM + h) * S_LEN + s) * D_DIM + d] = to_bf16_hw(v);
        } else {
          Vto[(((size_t)bidx * H_NUM + h) * D_DIM + d) * S_LEN + s] = to_bf16_hw(v);
        }
      }
}

// ---------------------------------------------------------------------------
// Kernel 2: flash attention.  One wave per 16-query tile; 64-key inner tile
// (16 WMMA per iteration) with online softmax; P relayout via per-wave LDS.
// ---------------------------------------------------------------------------
__global__ __launch_bounds__(128)
void attn_kernel(const __bf16* __restrict__ Qi, const __bf16* __restrict__ Ki,
                 const __bf16* __restrict__ Vti, __bf16* __restrict__ Oo) {
  __shared__ __align__(32) __bf16 Pl[4][16][64];  // per-wave 16x64 P tile

  const int tid = threadIdx.x, lane = tid & 31, wv = tid >> 5;
  const int wg = blockIdx.x * 4 + wv;     // 4096 q-tiles total
  const int bh = wg >> 7, qt = wg & 127;
  const int bI = bh >> 4, hI = bh & 15;
  const __bf16* Qbh = Qi  + (size_t)bh * S_LEN * D_DIM;
  const __bf16* Kbh = Ki  + (size_t)bh * S_LEN * D_DIM;
  const __bf16* Vbh = Vti + (size_t)bh * D_DIM * S_LEN;

  const int nl = lane & 15, uh = lane >> 4, hs = uh << 3;
  const float NEG_INF = -__builtin_inff();
  const float SC = 0.125f * 1.44269504088896341f;   // 1/sqrt(64) * log2(e)

  // Q A-fragments (resident for the whole KV loop)
  V16U qa[2];
  const __bf16* qrow = Qbh + (size_t)(qt * 16 + nl) * D_DIM;
#pragma unroll
  for (int c = 0; c < 2; ++c) {
    qa[c].h[0] = *(const v8bf*)(qrow + c * 32 + (uh << 4));
    qa[c].h[1] = *(const v8bf*)(qrow + c * 32 + (uh << 4) + 8);
  }

  float m_s[8], l_s[8];
  v8f oacc[4];
#pragma unroll
  for (int r = 0; r < 8; ++r) { m_s[r] = NEG_INF; l_s[r] = 0.0f; }
#pragma unroll
  for (int g = 0; g < 4; ++g)
#pragma unroll
    for (int e = 0; e < 8; ++e) oacc[g][e] = 0.0f;

  const int kend = qt * 16 + 16;          // causal: keys < kend
  for (int kt = 0; kt < kend; kt += 64) {
    // ---- scores: S = Q @ K^T  (four 16-key quarters, chained over D=64) ----
    v8f sacc[4];
#pragma unroll
    for (int t = 0; t < 4; ++t) {
#pragma unroll
      for (int e = 0; e < 8; ++e) sacc[t][e] = 0.0f;
      const __bf16* kp = Kbh + (size_t)(kt + t * 16 + nl) * D_DIM + (uh << 4);
      v16bf b0 = *(const v16bf*)kp;
      v16bf b1 = *(const v16bf*)(kp + 32);
      sacc[t] = __builtin_amdgcn_wmma_f32_16x16x32_bf16(
          false, qa[0].v, false, b0, (short)0, sacc[t], false, false);
      sacc[t] = __builtin_amdgcn_wmma_f32_16x16x32_bf16(
          false, qa[1].v, false, b1, (short)0, sacc[t], false, false);
    }
    // ---- causal mask + online softmax (row stats across 16-lane halves) ----
#pragma unroll
    for (int r = 0; r < 8; ++r) {
      int sq = qt * 16 + r + hs;
      float sv[4];
#pragma unroll
      for (int t = 0; t < 4; ++t) {
        sv[t] = sacc[t][r] * SC;
        if (kt + t * 16 + nl > sq) sv[t] = NEG_INF;
      }
      float mx = fmaxf(fmaxf(sv[0], sv[1]), fmaxf(sv[2], sv[3]));
#pragma unroll
      for (int off = 1; off < 16; off <<= 1)
        mx = fmaxf(mx, __shfl_xor(mx, off, 32));
      float mn = fmaxf(m_s[r], mx);
      float pv[4];
#pragma unroll
      for (int t = 0; t < 4; ++t) pv[t] = exp2f(sv[t] - mn);
      float al = exp2f(m_s[r] - mn);
      float rs = (pv[0] + pv[1]) + (pv[2] + pv[3]);
#pragma unroll
      for (int off = 1; off < 16; off <<= 1)
        rs += __shfl_xor(rs, off, 32);
      l_s[r] = l_s[r] * al + rs;
      m_s[r] = mn;
#pragma unroll
      for (int g = 0; g < 4; ++g) oacc[g][r] *= al;
#pragma unroll
      for (int t = 0; t < 4; ++t)
        Pl[wv][r + hs][t * 16 + nl] = to_bf16_hw(pv[t]);
    }
    // ---- P C-layout -> A-layout via LDS (DS ops in-order within a wave) ----
    V16U pa[2];
#pragma unroll
    for (int c = 0; c < 2; ++c) {
      pa[c].h[0] = *(const v8bf*)&Pl[wv][nl][c * 32 + (uh << 4)];
      pa[c].h[1] = *(const v8bf*)&Pl[wv][nl][c * 32 + (uh << 4) + 8];
    }
    // ---- O += P @ V  (V transposed: contiguous B fragments) ----
#pragma unroll
    for (int g = 0; g < 4; ++g) {
      const __bf16* vp = Vbh + (size_t)(g * 16 + nl) * S_LEN + kt + (uh << 4);
      v16bf vf0 = *(const v16bf*)vp;
      v16bf vf1 = *(const v16bf*)(vp + 32);
      oacc[g] = __builtin_amdgcn_wmma_f32_16x16x32_bf16(
          false, pa[0].v, false, vf0, (short)0, oacc[g], false, false);
      oacc[g] = __builtin_amdgcn_wmma_f32_16x16x32_bf16(
          false, pa[1].v, false, vf1, (short)0, oacc[g], false, false);
    }
  }

  // ---- finalize: O * (1/l), write bf16 attn_out [B,S,E] ----
  float rl[8];
#pragma unroll
  for (int r = 0; r < 8; ++r) rl[r] = __builtin_amdgcn_rcpf(l_s[r]);
#pragma unroll
  for (int g = 0; g < 4; ++g)
#pragma unroll
    for (int r = 0; r < 8; ++r) {
      int sq = qt * 16 + r + hs;
      int dg = g * 16 + nl;
      float o = oacc[g][r] * rl[r];
      Oo[((size_t)(bI * S_LEN + sq)) * E_DIM + hI * D_DIM + dg] = to_bf16_hw(o);
    }
}

// ---------------------------------------------------------------------------
// Kernel 3: out = attn_out(bf16) @ W_proj + b_proj, fp32 output.
// M=4096, N=1024, K=1024.  A tile (already bf16) is staged with
// GLOBAL_LOAD_ASYNC_TO_LDS_B128: each lane carries its own LDS destination
// offset, so the fragment-order swizzle is done by the async unit and the
// data never passes through VGPRs.  Completion via s_wait_asynccnt.
// ---------------------------------------------------------------------------
__global__ __launch_bounds__(256)
void proj_gemm_kernel(const __bf16* __restrict__ Ai, const float* __restrict__ W,
                      const float* __restrict__ bias, float* __restrict__ Co) {
  const int K = 1024, N = 1024;
  __shared__ __align__(32) __bf16 As[8][32][16];
  __shared__ __align__(32) __bf16 Bs[4][32][16];

  const int tid = threadIdx.x, lane = tid & 31, wv = tid >> 5;
  const int m0 = blockIdx.y * 128, n0 = blockIdx.x * 64;
  const int wrg = (wv >> 1) * 2, wcg = (wv & 1) * 2;

  // Per-lane async-copy task setup (two 16B chunks per thread per K-step).
  unsigned a_ldsoff[2];
  int a_mloc[2], a_kc[2];
#pragma unroll
  for (int t = 0; t < 2; ++t) {
    int task = tid + t * 256;
    a_mloc[t] = task >> 2;
    a_kc[t]   = (task & 3) << 3;
    int g = a_mloc[t] >> 4, row = a_mloc[t] & 15;
    int lane_s = (((a_kc[t] >> 3) & 1) << 4) + row;
    a_ldsoff[t] = lds_offset(&As[g][lane_s][(a_kc[t] >> 4) << 3]);
  }

  v8f acc[2][2];
#pragma unroll
  for (int i = 0; i < 2; ++i)
#pragma unroll
    for (int j = 0; j < 2; ++j)
#pragma unroll
      for (int e = 0; e < 8; ++e) acc[i][j][e] = 0.0f;

  for (int kt = 0; kt < K; kt += 32) {
    __syncthreads();
    // ---- stage A via async global->LDS (bf16, fragment-order scatter) ----
#pragma unroll
    for (int t = 0; t < 2; ++t) {
      const __bf16* src = Ai + (size_t)(m0 + a_mloc[t]) * K + kt + a_kc[t];
      asm volatile("global_load_async_to_lds_b128 %0, %1, off"
                   :: "v"(a_ldsoff[t]), "v"(src) : "memory");
    }
    // ---- stage B: 32x64 fp32 strided columns -> bf16 fragment-order LDS ----
    {
      int ncol = tid & 63, kq = tid >> 6;
      const float* wsrc = W + (size_t)(kt + (kq << 3)) * N + n0 + ncol;
      int lane_s = ((kq & 1) << 4) + (ncol & 15);
      int cg = ncol >> 4, off = (kq >> 1) << 3;
      float f[8];
#pragma unroll
      for (int j2 = 0; j2 < 8; ++j2) f[j2] = wsrc[(size_t)j2 * N];
      v4u w;
#pragma unroll
      for (int j2 = 0; j2 < 4; ++j2) w[j2] = cvt_pk_bf16(f[2 * j2], f[2 * j2 + 1]);
      *(v4u*)&Bs[cg][lane_s][off] = w;
    }
    asm volatile("s_wait_asynccnt 0" ::: "memory");
    __syncthreads();
    v16bf af[2], bfr[2];
    af[0]  = *(const v16bf*)&As[wrg][lane][0];
    af[1]  = *(const v16bf*)&As[wrg + 1][lane][0];
    bfr[0] = *(const v16bf*)&Bs[wcg][lane][0];
    bfr[1] = *(const v16bf*)&Bs[wcg + 1][lane][0];
#pragma unroll
    for (int i = 0; i < 2; ++i)
#pragma unroll
      for (int j = 0; j < 2; ++j)
        acc[i][j] = __builtin_amdgcn_wmma_f32_16x16x32_bf16(
            false, af[i], false, bfr[j], (short)0, acc[i][j], false, false);
  }

  const int rowbase = (lane >> 4) << 3;
  const int ncl = lane & 15;
#pragma unroll
  for (int i = 0; i < 2; ++i)
#pragma unroll
    for (int j = 0; j < 2; ++j)
#pragma unroll
      for (int r = 0; r < 8; ++r) {
        int mg = m0 + ((wrg + i) << 4) + r + rowbase;
        int ng = n0 + ((wcg + j) << 4) + ncl;
        Co[(size_t)mg * N + ng] = acc[i][j][r] + bias[ng];
      }
}

// ---------------------------------------------------------------------------
extern "C" void kernel_launch(void* const* d_in, const int* in_sizes, int n_in,
                              void* d_out, int out_size, void* d_ws,
                              size_t ws_size, hipStream_t stream) {
  (void)in_sizes; (void)n_in; (void)out_size; (void)ws_size;
  const float* x      = (const float*)d_in[0];
  const float* W_attn = (const float*)d_in[1];
  const float* b_attn = (const float*)d_in[2];
  const float* W_proj = (const float*)d_in[3];
  const float* b_proj = (const float*)d_in[4];
  float* out = (float*)d_out;

  const size_t nQ = (size_t)2 * H_NUM * S_LEN * D_DIM;   // 4M elements
  __bf16* q  = (__bf16*)d_ws;
  __bf16* k  = q + nQ;
  __bf16* vt = k + nQ;
  __bf16* ao = vt + nQ;                                   // total 32 MB

  qkv_gemm_kernel<<<dim3(48, 32), 256, 0, stream>>>(x, W_attn, b_attn, q, k, vt);
  attn_kernel<<<dim3(1024), 128, 0, stream>>>(q, k, vt, ao);
  proj_gemm_kernel<<<dim3(16, 32), 256, 0, stream>>>(ao, W_proj, b_proj, out);
}
